// MVCMGNet_37434934952592
// MI455X (gfx1250) — matrix operand order repaired and checked by reading, not compiled, not run on hardware
//
#include <hip/hip_runtime.h>
#include <hip/hip_bf16.h>

#define G_ 4
#define B_ 128
#define N_ 30
#define F_ 5000
#define C_ 10
#define T_ 8
#define ROWS (B_*N_)          // 3840 rows per band
#define KCH ((F_+31)/32)      // 157 K-chunks of 32 (156 full + 1 tail of 8)
#define KFULL (F_/32)         // 156 full chunks

typedef __attribute__((ext_vector_type(16))) _Float16 v16h;
typedef __attribute__((ext_vector_type(8)))  float    v8f;

// ---------------------------------------------------------------------------
// Kernel 0: cor[g,i,j] = mean_t cmc_train[t,g,i,j]
// ---------------------------------------------------------------------------
__global__ void cor_kernel(const float* __restrict__ cmc_train,
                           float* __restrict__ cor) {
  int idx = blockIdx.x * 256 + threadIdx.x;
  if (idx < G_*N_*N_) {
    float s = 0.f;
    #pragma unroll
    for (int t = 0; t < T_; ++t) s += cmc_train[t*G_*N_*N_ + idx];
    cor[idx] = s * (1.0f / T_);
  }
}

// ---------------------------------------------------------------------------
// Kernel 1: pre-transform W into WMMA B-fragment lane layout (f16, global).
// Fragment element (chunk c, lane ln, half j2) = W[c*32 + (ln>>4)*16 + j2,
// ln&15], zero-padded for k>=F or n>=C.  One (c,ln) pair = 16 contiguous k
// -> 16 independent loads batched behind one wait, one 32B vector store.
// ---------------------------------------------------------------------------
__global__ void __launch_bounds__(256) wprep_kernel(
    const float* __restrict__ W_eeg, const float* __restrict__ W_emg,
    const float* __restrict__ W_cmc, _Float16* __restrict__ wfrag)
{
  const int g = blockIdx.x, mod = blockIdx.y;
  const float* W = (mod == 0) ? W_eeg : (mod == 1) ? W_emg : W_cmc;
  const float* Wg = W + g * F_ * C_;
  _Float16* out = wfrag + (size_t)(mod * G_ + g) * KCH * 512;
  for (int p = threadIdx.x; p < KCH * 32; p += 256) {   // p = c*32 + ln
    int ln = p & 31;
    int c  = p >> 5;
    int nn = ln & 15, hi2 = ln >> 4;
    int kbase = c * 32 + hi2 * 16;
    int ncl = nn < C_ ? nn : 0;
    float vals[16];
    #pragma unroll
    for (int j = 0; j < 16; ++j) {
      int kk = kbase + j;
      int kc = kk < F_ ? kk : F_ - 1;
      vals[j] = Wg[kc * C_ + ncl];
    }
    v16h vb;
    #pragma unroll
    for (int j = 0; j < 16; ++j) {
      int kk = kbase + j;
      float m = (kk < F_ && nn < C_) ? 1.0f : 0.0f;
      vb[j] = (_Float16)(vals[j] * m);
    }
    *(v16h*)&out[(size_t)p * 16] = vb;
  }
}

// ---------------------------------------------------------------------------
// Kernel 2: heavy projection  h[mod][g][row][c] = x[g,row,:] @ W[g,:,c]
// WMMA f32<-f16, 16x16x32.  grid = (G, 30 row-tiles-of-8-waves, 3 modalities)
// Pure stream: per chunk 4x global_load_b128 (A, HBM) + 2x global_load_b128
// (B fragment, L2-resident) + 1 wmma.  No LDS, no barriers.
// ---------------------------------------------------------------------------
__global__ void __launch_bounds__(256) proj_wmma_kernel(
    const float* __restrict__ eeg, const float* __restrict__ emg,
    const float* __restrict__ cmc,
    const _Float16* __restrict__ wfrag,
    float* __restrict__ h_all)
{
  const int g = blockIdx.x, ytile = blockIdx.y, mod = blockIdx.z;
  const float* x = (mod == 0) ? eeg : (mod == 1) ? emg : cmc;
  const float* xg = x + (size_t)g * B_ * N_ * F_;
  const int tid  = threadIdx.x;
  const int wave = tid >> 5, lane = tid & 31;
  const int r = lane & 15, hi = lane >> 4;
  const int m0 = (ytile * 8 + wave) * 16;               // tile base row (0..3824)
  // A order used here (contiguous, same as wfrag): lanes 0-15 hold K {0..15}, lanes 16-31 {16..31}
  const float* xrow  = xg + (size_t)(m0 + r) * F_;      // this lane's A row
  const float* xbase = xrow + hi * 16;
  const _Float16* wf = wfrag + (size_t)(mod * G_ + g) * KCH * 512 + lane * 16;
  v8f acc = {};
  #pragma unroll 4
  for (int cc = 0; cc < KFULL; ++cc) {
    const float* p = xbase + cc * 32;
    float4 a0 = *(const float4*)(p);
    float4 a1 = *(const float4*)(p + 4);
    float4 a2 = *(const float4*)(p + 8);
    float4 a3 = *(const float4*)(p + 12);
    v16h b = *(const v16h*)(wf + (size_t)cc * 512);
    v16h a;
    a[0]=(_Float16)a0.x;  a[1]=(_Float16)a0.y;  a[2]=(_Float16)a0.z;  a[3]=(_Float16)a0.w;
    a[4]=(_Float16)a1.x;  a[5]=(_Float16)a1.y;  a[6]=(_Float16)a1.z;  a[7]=(_Float16)a1.w;
    a[8]=(_Float16)a2.x;  a[9]=(_Float16)a2.y;  a[10]=(_Float16)a2.z; a[11]=(_Float16)a2.w;
    a[12]=(_Float16)a3.x; a[13]=(_Float16)a3.y; a[14]=(_Float16)a3.z; a[15]=(_Float16)a3.w;
    acc = __builtin_amdgcn_wmma_f32_16x16x32_f16(false, a, false, b,
                                                 (short)0, acc, false, false);
  }
  {
    // Tail chunk (K = 4992..4999): only lanes hi==0 hold valid A data.
    v16h a = {};
    if (hi == 0) {
      float4 a0 = *(const float4*)(xrow + KFULL * 32);
      float4 a1 = *(const float4*)(xrow + KFULL * 32 + 4);
      a[0]=(_Float16)a0.x; a[1]=(_Float16)a0.y; a[2]=(_Float16)a0.z; a[3]=(_Float16)a0.w;
      a[4]=(_Float16)a1.x; a[5]=(_Float16)a1.y; a[6]=(_Float16)a1.z; a[7]=(_Float16)a1.w;
    }
    v16h b = *(const v16h*)(wf + (size_t)KFULL * 512);
    acc = __builtin_amdgcn_wmma_f32_16x16x32_f16(false, a, false, b,
                                                 (short)0, acc, false, false);
  }
  // C/D layout: vgpr i, lanes 0-15 -> (M=i, N=lane); lanes 16-31 -> (M=8+i, N=lane-16)
  if (r < C_) {
    float* hout = h_all + (size_t)(mod * G_ + g) * ROWS * C_;
    #pragma unroll
    for (int i = 0; i < 8; ++i) {
      int M = i + hi * 8;
      hout[(size_t)(m0 + M) * C_ + r] = acc[i];
    }
  }
}

// ---------------------------------------------------------------------------
// Kernel 3: per (g,b): GAT attention (eeg,emg) + CMC GCN + gating0 -> fused
// ---------------------------------------------------------------------------
__global__ void __launch_bounds__(128) fuse_gat_kernel(
    const float* __restrict__ h_all, const float* __restrict__ cor,
    const float* __restrict__ adj_eeg, const float* __restrict__ adj_emg,
    const float* __restrict__ a_eeg, const float* __restrict__ a_emg,
    const float* __restrict__ b_cmc,
    const float* __restrict__ g0_w1, const float* __restrict__ g0_b1,
    const float* __restrict__ g0_w2, const float* __restrict__ g0_b2,
    float* __restrict__ fused)
{
  const int gb = blockIdx.x;
  const int g = gb / B_, b = gb % B_;
  const int t = threadIdx.x;
  __shared__ float hbuf[3][N_ * C_];
  __shared__ float att[N_ * N_];
  __shared__ float f1s[N_], f2s[N_];
  __shared__ float outb[3][N_ * C_];
  for (int m = 0; m < 3; ++m) {
    const float* src = h_all + ((size_t)(m * G_ + g) * ROWS + (size_t)b * N_) * C_;
    for (int idx = t; idx < N_ * C_; idx += 128) hbuf[m][idx] = src[idx];
  }
  __syncthreads();
  for (int m = 0; m < 2; ++m) {
    const float* h   = hbuf[m];
    const float* a   = (m == 0 ? a_eeg : a_emg) + g * 2 * C_;
    const float* adj = (m == 0 ? adj_eeg : adj_emg) + g * N_ * N_;
    if (t < N_) {
      float s1 = 0.f, s2 = 0.f;
      #pragma unroll
      for (int c = 0; c < C_; ++c) { s1 += h[t*C_+c]*a[c]; s2 += h[t*C_+c]*a[C_+c]; }
      f1s[t] = s1; f2s[t] = s2;
    }
    __syncthreads();
    if (t < N_) {                          // softmax over i (axis=2) per column j
      int j = t;
      float ecol[N_];
      float mx = -3.0e38f;
      #pragma unroll
      for (int i = 0; i < N_; ++i) {
        float e = f1s[i] + f2s[j];
        e = (e > 0.f) ? e : 0.1f * e;      // LeakyReLU alpha=0.1
        if (!(adj[i*N_+j] > 0.f)) e = -1.0e12f;
        ecol[i] = e; mx = fmaxf(mx, e);
      }
      float ssum = 0.f;
      #pragma unroll
      for (int i = 0; i < N_; ++i) { ecol[i] = __expf(ecol[i] - mx); ssum += ecol[i]; }
      float inv = 1.f / ssum;
      #pragma unroll
      for (int i = 0; i < N_; ++i) att[i*N_+j] = ecol[i] * inv;
    }
    __syncthreads();
    for (int idx = t; idx < N_ * C_; idx += 128) {   // out[i,d] = sum_j att[i,j] h[j,d]
      int i = idx / C_, d = idx % C_;
      float s = 0.f;
      #pragma unroll
      for (int j = 0; j < N_; ++j) s += att[i*N_+j] * h[j*C_+d];
      outb[m][idx] = fmaxf(s, 0.f);
    }
    __syncthreads();
  }
  {  // CMC GCN: cor[g] @ hc + b_cmc
    const float* hc = hbuf[2];
    const float* cg = cor + g * N_ * N_;
    for (int idx = t; idx < N_ * C_; idx += 128) {
      int i = idx / C_, d = idx % C_;
      float s = b_cmc[g*C_ + d];
      #pragma unroll
      for (int j = 0; j < N_; ++j) s += cg[i*N_+j] * hc[j*C_+d];
      outb[2][idx] = s;
    }
  }
  __syncthreads();
  if (t < N_) {  // gating0 per node: 30 -> 64 relu -> 3 softmax
    float in[3 * C_];
    #pragma unroll
    for (int e = 0; e < 3; ++e)
      #pragma unroll
      for (int d = 0; d < C_; ++d) in[e*C_+d] = outb[e][t*C_+d];
    float logit[3] = {g0_b2[0], g0_b2[1], g0_b2[2]};
    for (int hh = 0; hh < 64; ++hh) {
      float s = g0_b1[hh];
      #pragma unroll
      for (int f = 0; f < 30; ++f) s += in[f] * g0_w1[f*64+hh];
      s = fmaxf(s, 0.f);
      logit[0] += s * g0_w2[hh*3+0];
      logit[1] += s * g0_w2[hh*3+1];
      logit[2] += s * g0_w2[hh*3+2];
    }
    float mx = fmaxf(logit[0], fmaxf(logit[1], logit[2]));
    float p0 = __expf(logit[0]-mx), p1 = __expf(logit[1]-mx), p2 = __expf(logit[2]-mx);
    float inv = 1.f / (p0 + p1 + p2);
    p0 *= inv; p1 *= inv; p2 *= inv;
    float* fo = fused + ((size_t)(g * B_ + b) * N_ + t) * C_;
    #pragma unroll
    for (int d = 0; d < C_; ++d)
      fo[d] = in[d]*p0 + in[C_+d]*p1 + in[2*C_+d]*p2;
  }
}

// ---------------------------------------------------------------------------
// Kernel 4: per b: band gating1 + node-mean pool + classifier head -> (B,4)
// ---------------------------------------------------------------------------
__global__ void __launch_bounds__(64) head_kernel(
    const float* __restrict__ fused,
    const float* __restrict__ g1_w1, const float* __restrict__ g1_b1,
    const float* __restrict__ g1_w2, const float* __restrict__ g1_b2,
    const float* __restrict__ mlp0_w, const float* __restrict__ mlp0_b,
    const float* __restrict__ mlp1_w, const float* __restrict__ mlp1_b,
    const float* __restrict__ mlp2_w, const float* __restrict__ mlp2_b,
    const float* __restrict__ bn_g, const float* __restrict__ bn_b,
    float* __restrict__ out)
{
  const int b = blockIdx.x, t = threadIdx.x;
  __shared__ float band[N_ * 40];
  __shared__ float onodes[N_ * 40];
  __shared__ float feat[40];
  __shared__ float x1[256];
  __shared__ float x2[32];
  for (int idx = t; idx < N_ * 40; idx += 64) {   // band[i][g*10+d]
    int i = idx / 40, q = idx % 40, g = q / 10, d = q % 10;
    band[idx] = fused[((size_t)(g * B_ + b) * N_ + i) * C_ + d];
  }
  __syncthreads();
  if (t < N_) {  // gating1 per node: 40 -> 64 relu -> 4 softmax, scale bands
    float inp[40];
    #pragma unroll
    for (int q = 0; q < 40; ++q) inp[q] = band[t*40+q];
    float logit[4] = {g1_b2[0], g1_b2[1], g1_b2[2], g1_b2[3]};
    for (int hh = 0; hh < 64; ++hh) {
      float s = g1_b1[hh];
      #pragma unroll
      for (int f = 0; f < 40; ++f) s += inp[f] * g1_w1[f*64+hh];
      s = fmaxf(s, 0.f);
      #pragma unroll
      for (int k = 0; k < 4; ++k) logit[k] += s * g1_w2[hh*4+k];
    }
    float mx = fmaxf(fmaxf(logit[0], logit[1]), fmaxf(logit[2], logit[3]));
    float p[4]; float ssum = 0.f;
    #pragma unroll
    for (int k = 0; k < 4; ++k) { p[k] = __expf(logit[k]-mx); ssum += p[k]; }
    float inv = 1.f / ssum;
    #pragma unroll
    for (int q = 0; q < 40; ++q) onodes[t*40+q] = inp[q] * (p[q/10] * inv);
  }
  __syncthreads();
  if (t < 40) {   // deterministic serial node-mean
    float s = 0.f;
    for (int i = 0; i < N_; ++i) s += onodes[i*40+t];
    feat[t] = s * (1.0f / N_);
  }
  __syncthreads();
  for (int o = t; o < 256; o += 64) {
    float s = mlp0_b[o];
    #pragma unroll
    for (int f = 0; f < 40; ++f) s += feat[f] * mlp0_w[f*256+o];
    x1[o] = fmaxf(s, 0.f);
  }
  __syncthreads();
  if (t < 32) {
    float s = mlp1_b[t];
    for (int f = 0; f < 256; ++f) s += x1[f] * mlp1_w[f*32+t];
    s = fmaxf(s, 0.f);
    s = s * (bn_g[t] * rsqrtf(1.0f + 1e-5f)) + bn_b[t];  // BN eval, mean0/var1
    x2[t] = s;
  }
  __syncthreads();
  if (t < 4) {
    float s = mlp2_b[t];
    #pragma unroll
    for (int f = 0; f < 32; ++f) s += x2[f] * mlp2_w[f*4+t];
    out[b*4 + t] = s;
  }
}

// ---------------------------------------------------------------------------
extern "C" void kernel_launch(void* const* d_in, const int* in_sizes, int n_in,
                              void* d_out, int out_size, void* d_ws, size_t ws_size,
                              hipStream_t stream) {
  const float* eeg      = (const float*)d_in[0];
  const float* wpli_eeg = (const float*)d_in[1];
  const float* emg      = (const float*)d_in[2];
  const float* wpli_emg = (const float*)d_in[3];
  const float* cmc      = (const float*)d_in[4];
  const float* cmc_tr   = (const float*)d_in[5];
  const float* W_eeg = (const float*)d_in[6];
  const float* a_eeg = (const float*)d_in[7];
  const float* W_emg = (const float*)d_in[8];
  const float* a_emg = (const float*)d_in[9];
  const float* W_cmc = (const float*)d_in[10];
  const float* b_cmc = (const float*)d_in[11];
  const float* g0_w1 = (const float*)d_in[12];
  const float* g0_b1 = (const float*)d_in[13];
  const float* g0_w2 = (const float*)d_in[14];
  const float* g0_b2 = (const float*)d_in[15];
  const float* g1_w1 = (const float*)d_in[16];
  const float* g1_b1 = (const float*)d_in[17];
  const float* g1_w2 = (const float*)d_in[18];
  const float* g1_b2 = (const float*)d_in[19];
  const float* mlp0_w = (const float*)d_in[20];
  const float* mlp0_b = (const float*)d_in[21];
  const float* mlp1_w = (const float*)d_in[22];
  const float* mlp1_b = (const float*)d_in[23];
  const float* mlp2_w = (const float*)d_in[24];
  const float* mlp2_b = (const float*)d_in[25];
  const float* bn_g = (const float*)d_in[26];
  const float* bn_b = (const float*)d_in[27];

  float* ws    = (float*)d_ws;
  float* h_all = ws;                               // 460800 floats
  float* cor   = h_all + 3 * G_ * ROWS * C_;       // 3600 floats
  float* fused = cor + G_ * N_ * N_;               // 153600 floats
  _Float16* wfrag = (_Float16*)(fused + (size_t)G_ * B_ * N_ * C_);  // 964608 halfs

  hipLaunchKernelGGL(cor_kernel, dim3((G_*N_*N_ + 255) / 256), dim3(256), 0, stream,
                     cmc_tr, cor);
  hipLaunchKernelGGL(wprep_kernel, dim3(G_, 3), dim3(256), 0, stream,
                     W_eeg, W_emg, W_cmc, wfrag);
  hipLaunchKernelGGL(proj_wmma_kernel, dim3(G_, 30, 3), dim3(256), 0, stream,
                     eeg, emg, cmc, wfrag, h_all);
  hipLaunchKernelGGL(fuse_gat_kernel, dim3(G_ * B_), dim3(128), 0, stream,
                     h_all, cor, wpli_eeg, wpli_emg, a_eeg, a_emg, b_cmc,
                     g0_w1, g0_b1, g0_w2, g0_b2, fused);
  hipLaunchKernelGGL(head_kernel, dim3(B_), dim3(64), 0, stream,
                     fused, g1_w1, g1_b1, g1_w2, g1_b2,
                     mlp0_w, mlp0_b, mlp1_w, mlp1_b, mlp2_w, mlp2_b,
                     bn_g, bn_b, (float*)d_out);
}
